// LearnableMultiheadSelfAttention_50517405335635
// MI455X (gfx1250) — matrix operand order repaired and checked by reading, not compiled, not run on hardware
//
#include <hip/hip_runtime.h>
#include <hip/hip_bf16.h>

#define LQ   512
#define BZ   4
#define DM   512
#define NH   8
#define DHD  64
#define MEMN 3
#define TOT  2048        // (MEMN+1)*LQ
#define HD   512         // NH*DHD
#define BH   32          // BZ*NH
#define ATT_SCALE 0.125f // 1/sqrt(64)

typedef _Float16 v8h  __attribute__((ext_vector_type(8)));
typedef _Float16 v16h __attribute__((ext_vector_type(16)));
typedef float    v8f  __attribute__((ext_vector_type(8)));

// ---------------------------------------------------------------------------
// f32 -> f16 copy/convert
// ---------------------------------------------------------------------------
__global__ void cvt_f32_to_f16(const float* __restrict__ s, _Float16* __restrict__ d, int n) {
  int i = blockIdx.x * 256 + threadIdx.x;
  if (i < n) d[i] = (_Float16)s[i];
}

// Transpose + convert a weight block: WT[n*K + k] = W[k*ldw + colOff + n]  (f32 -> f16)
__global__ void wtrans_f16(const float* __restrict__ W, _Float16* __restrict__ WT,
                           int K, int N, int ldw, int colOff) {
  int idx = blockIdx.x * 256 + threadIdx.x;
  if (idx >= K * N) return;
  int n = idx / K;
  int k = idx - n * K;
  WT[idx] = (_Float16)W[(long long)k * ldw + colOff + n];
}

// qu/qv = q + pos_bias  (f32 -> f16), layout [i*B+b, HD]
__global__ void qbias_f16(const float* __restrict__ q, const float* __restrict__ bu,
                          const float* __restrict__ bv, _Float16* __restrict__ qu,
                          _Float16* __restrict__ qv, int n) {
  int i = blockIdx.x * 256 + threadIdx.x;
  if (i >= n) return;
  int d = i & (HD - 1);
  float qv32 = q[i];
  qu[i] = (_Float16)(qv32 + bu[d]);
  qv[i] = (_Float16)(qv32 + bv[d]);
}

// VT[bn][d][c] = V[(c*B + b)*HD + n*64 + d]
__global__ void vtrans_f16(const _Float16* __restrict__ V, _Float16* __restrict__ VT) {
  int idx = blockIdx.x * 256 + threadIdx.x;     // over BH*DHD*TOT = 4M
  int bn  = idx >> 17;                          // / (64*2048)
  int rem = idx & 131071;
  int d   = rem >> 11;
  int c   = rem & 2047;
  int b = bn >> 3, n = bn & 7;
  VT[idx] = V[(long long)(c * BZ + b) * HD + n * DHD + d];
}

// ---------------------------------------------------------------------------
// Fragment loaders (VGPR layouts per CDNA5 ISA §7.12.2):
//  A: lanes 0-15 hold K 0-7 / 16-23, lanes 16-31 hold K 8-15 / 24-31
//  B: same as A: lanes 0-15 hold K 0-7 / 16-23, lanes 16-31 hold K 8-15 / 24-31
// ---------------------------------------------------------------------------
__device__ __forceinline__ v16h load_frag_a(const _Float16* p, int kb, int half_) {
  v8h lo = *(const v8h*)(p + kb + 8 * half_);
  v8h hi = *(const v8h*)(p + kb + 16 + 8 * half_);
  v16h r;
#pragma unroll
  for (int t = 0; t < 8; ++t) { r[t] = lo[t]; r[t + 8] = hi[t]; }
  return r;
}
__device__ __forceinline__ v16h load_frag_b(const _Float16* p, int kb, int half_) {
  v8h lo = *(const v8h*)(p + kb + 8 * half_);
  v8h hi = *(const v8h*)(p + kb + 16 + 8 * half_);
  v16h r;
#pragma unroll
  for (int t = 0; t < 8; ++t) { r[t] = lo[t]; r[t + 8] = hi[t]; }
  return r;
}

// ---------------------------------------------------------------------------
// Batched NT GEMM: C[M,N] = A[M,K] * B[N,K]^T, f16 in, f32 WMMA accumulate,
// OutT in {float, _Float16}. 2x4 register blocking: each wave owns a 32x64
// tile (8 accumulators); A fragments reused 4x, B fragments 2x.
// 8 waves per block; tile guard is wave-uniform so EXEC stays all-ones.
// Requires M % 32 == 0, N % 64 == 0, K % 32 == 0 (true for every call here).
// ---------------------------------------------------------------------------
template <typename OutT>
__global__ void gemm_nt_wmma(const _Float16* __restrict__ A, const _Float16* __restrict__ B,
                             OutT* __restrict__ C, int M, int N, int K,
                             int lda, int ldb, int ldc,
                             long long bsA, long long bsB, long long bsC) {
  int lane = threadIdx.x & 31;
  int wave = threadIdx.x >> 5;
  int tile = blockIdx.x * (blockDim.x >> 5) + wave;
  int tilesN = N >> 6;                          // 64-wide tiles
  int tilesM = M >> 5;                          // 32-tall tiles
  if (tile >= tilesM * tilesN) return;          // uniform per wave
  int m0 = (tile / tilesN) << 5;
  int n0 = (tile % tilesN) << 6;

  long long batch = blockIdx.y;
  A += batch * bsA;  B += batch * bsB;  C += batch * bsC;

  int half_ = lane >> 4;                        // 0/1 half-wave
  int sub   = lane & 15;
  const _Float16* ap0 = A + (long long)(m0 + sub) * lda;
  const _Float16* ap1 = A + (long long)(m0 + 16 + sub) * lda;
  const _Float16* bp  = B + (long long)(n0 + sub) * ldb;
  const long long bstep = (long long)16 * ldb;  // next 16-column group of B^T

  v8f acc[2][4] = {};
  for (int kb = 0; kb < K; kb += 32) {
    v16h a0 = load_frag_a(ap0, kb, half_);
    v16h a1 = load_frag_a(ap1, kb, half_);
    v16h bf[4];
#pragma unroll
    for (int j = 0; j < 4; ++j) bf[j] = load_frag_b(bp + j * bstep, kb, half_);
#pragma unroll
    for (int j = 0; j < 4; ++j)
      acc[0][j] = __builtin_amdgcn_wmma_f32_16x16x32_f16(false, a0, false, bf[j],
                                                         (short)0, acc[0][j], false, false);
#pragma unroll
    for (int j = 0; j < 4; ++j)
      acc[1][j] = __builtin_amdgcn_wmma_f32_16x16x32_f16(false, a1, false, bf[j],
                                                         (short)0, acc[1][j], false, false);
  }
  // D: VGPR r -> row r (+8 for upper half-wave), lane&15 -> column
#pragma unroll
  for (int mi = 0; mi < 2; ++mi) {
#pragma unroll
    for (int j = 0; j < 4; ++j) {
#pragma unroll
      for (int r = 0; r < 8; ++r) {
        int row = m0 + mi * 16 + r + 8 * half_;
        C[(long long)row * ldc + (n0 + j * 16 + sub)] = (OutT)acc[mi][j][r];
      }
    }
  }
}

// ---------------------------------------------------------------------------
// Fused rel-shift + gate + causal mask + softmax over 2048 keys.
// Block = one (bn, i) row; 256 threads x 8 keys. prob written f16 [bn][i][c].
// Rel-shift closed form: BD_shifted[i,c] = BD_raw[i, c+511-i]; the shift's
// wrap region is exactly the causal mask c > 1536+i.
// ---------------------------------------------------------------------------
__global__ void softmax_kernel(const float* __restrict__ AC, const float* __restrict__ BD,
                               const float* __restrict__ ib, _Float16* __restrict__ prob) {
  int i  = blockIdx.x;   // 0..511
  int bn = blockIdx.y;   // 0..31
  int b  = bn >> 3;
  const float* ac = AC + ((long long)bn * LQ + i) * TOT;
  const float* bd = BD + ((long long)bn * LQ + i) * TOT;
  __shared__ float red[256];
  int t = threadIdx.x;
  float s[8];
  float mx = -__builtin_inff();
#pragma unroll
  for (int u = 0; u < 8; ++u) {
    int c = u * 256 + t;
    float v;
    if (c > 3 * LQ + i) {                 // causal mask == rel_shift wrap region
      v = -__builtin_inff();
    } else {
      int cp = c + (LQ - 1) - i;          // pre-shift index, in [0, 2047]
      float gA = ib[i * 16 + b * 4 + (c >> 9)];
      float gB = ib[i * 16 + b * 4 + (cp >> 9)];
      v = ATT_SCALE * (gA * ac[c] + gB * bd[cp]);
    }
    s[u] = v;
    mx = fmaxf(mx, v);
  }
  red[t] = mx; __syncthreads();
  for (int o = 128; o; o >>= 1) { if (t < o) red[t] = fmaxf(red[t], red[t + o]); __syncthreads(); }
  mx = red[0]; __syncthreads();
  float sum = 0.f;
#pragma unroll
  for (int u = 0; u < 8; ++u) {
    float e = (s[u] == -__builtin_inff()) ? 0.f : __expf(s[u] - mx);
    s[u] = e; sum += e;
  }
  red[t] = sum; __syncthreads();
  for (int o = 128; o; o >>= 1) { if (t < o) red[t] += red[t + o]; __syncthreads(); }
  float inv = 1.f / red[0];
  _Float16* pr = prob + ((long long)bn * LQ + i) * TOT;
#pragma unroll
  for (int u = 0; u < 8; ++u) pr[u * 256 + t] = (_Float16)(s[u] * inv);
}

// attn_matrix[i,c] = mean over 32 (b,n) slices of prob
__global__ void attn_mean_kernel(const _Float16* __restrict__ prob, float* __restrict__ out2) {
  int idx = blockIdx.x * 256 + threadIdx.x;    // over LQ*TOT = 1M
  float sum = 0.f;
#pragma unroll
  for (int bn = 0; bn < BH; ++bn) sum += (float)prob[(long long)bn * (LQ * TOT) + idx];
  out2[idx] = sum * (1.0f / 32.0f);
}

// y = x + attn_out; LayerNorm over D. One block per row (i*B+b).
__global__ void resid_ln_kernel(const float* __restrict__ x, const float* __restrict__ ao,
                                const float* __restrict__ g, const float* __restrict__ be,
                                float* __restrict__ out) {
  __shared__ float red[256];
  int row = blockIdx.x, t = threadIdx.x;
  const float* xr = x + (long long)row * DM;
  const float* ar = ao + (long long)row * DM;
  float y0 = xr[t] + ar[t];
  float y1 = xr[t + 256] + ar[t + 256];
  red[t] = y0 + y1; __syncthreads();
  for (int o = 128; o; o >>= 1) { if (t < o) red[t] += red[t + o]; __syncthreads(); }
  float mu = red[0] * (1.0f / 512.0f); __syncthreads();
  float d0 = y0 - mu, d1 = y1 - mu;
  red[t] = d0 * d0 + d1 * d1; __syncthreads();
  for (int o = 128; o; o >>= 1) { if (t < o) red[t] += red[t + o]; __syncthreads(); }
  float rstd = rsqrtf(red[0] * (1.0f / 512.0f) + 1e-5f);
  float* orow = out + (long long)row * DM;
  orow[t]       = d0 * rstd * g[t] + be[t];
  orow[t + 256] = d1 * rstd * g[t + 256] + be[t + 256];
}

// ---------------------------------------------------------------------------
extern "C" void kernel_launch(void* const* d_in, const int* in_sizes, int n_in,
                              void* d_out, int out_size, void* d_ws, size_t ws_size,
                              hipStream_t stream) {
  const float* x       = (const float*)d_in[0];   // [L,B,D]
  const float* memory  = (const float*)d_in[1];   // [MEM,L,B,D]
  const float* pos_emb = (const float*)d_in[2];   // [TOT,B,D]
  const float* pbu     = (const float*)d_in[3];   // [H,DH]
  const float* pbv     = (const float*)d_in[4];   // [H,DH]
  const float* indice  = (const float*)d_in[6];   // [L,B,4]
  const float* W_q     = (const float*)d_in[7];   // [D,512]
  const float* W_kv    = (const float*)d_in[8];   // [D,1024]
  const float* W_rel   = (const float*)d_in[9];   // [D,512]
  const float* W_o     = (const float*)d_in[10];  // [512,D]
  const float* ln_g    = (const float*)d_in[11];
  const float* ln_b    = (const float*)d_in[12];

  float* out1 = (float*)d_out;                 // [L,B,D]
  float* out2 = out1 + (size_t)LQ * BZ * DM;   // [L,TOT]

  // workspace carve-up (256B aligned)
  char* ws = (char*)d_ws;
  size_t off = 0;
  auto alloc = [&](size_t bytes) -> void* {
    void* p = ws + off;
    off = (off + bytes + 255) & ~(size_t)255;
    return p;
  };
  _Float16* hC    = (_Float16*)alloc((size_t)TOT * BZ * DM * 2);   // concat [mem;x], f16
  _Float16* hP    = (_Float16*)alloc((size_t)TOT * BZ * DM * 2);   // pos_emb f16
  _Float16* hWkT  = (_Float16*)alloc((size_t)HD * DM * 2);
  _Float16* hWvT  = (_Float16*)alloc((size_t)HD * DM * 2);
  _Float16* hWrT  = (_Float16*)alloc((size_t)HD * DM * 2);
  _Float16* hWqT  = (_Float16*)alloc((size_t)HD * DM * 2);
  _Float16* hWoT  = (_Float16*)alloc((size_t)DM * HD * 2);
  _Float16* hK    = (_Float16*)alloc((size_t)TOT * BZ * HD * 2);
  _Float16* hV    = (_Float16*)alloc((size_t)TOT * BZ * HD * 2);
  _Float16* hRel  = (_Float16*)alloc((size_t)TOT * BZ * HD * 2);
  float*    fQ    = (float*)   alloc((size_t)LQ * BZ * HD * 4);
  _Float16* hQu   = (_Float16*)alloc((size_t)LQ * BZ * HD * 2);
  _Float16* hQv   = (_Float16*)alloc((size_t)LQ * BZ * HD * 2);
  float*    fAC   = (float*)   alloc((size_t)BH * LQ * TOT * 4);   // 128 MB
  float*    fBD   = (float*)   alloc((size_t)BH * LQ * TOT * 4);   // 128 MB
  _Float16* hProb = (_Float16*)alloc((size_t)BH * LQ * TOT * 2);   // 64 MB
  _Float16* hVT   = (_Float16*)alloc((size_t)BH * DHD * TOT * 2);
  _Float16* hVec  = (_Float16*)alloc((size_t)LQ * BZ * HD * 2);
  float*    fAO   = (float*)   alloc((size_t)LQ * BZ * DM * 4);
  (void)ws_size; (void)in_sizes; (void)n_in; (void)out_size;

  const int memN = MEMN * LQ * BZ * DM;  // 3,145,728
  const int xN   = LQ * BZ * DM;         // 1,048,576
  const int posN = TOT * BZ * DM;        // 4,194,304

  // 1) build f16 concat c = [memory ; x] and pos_emb
  cvt_f32_to_f16<<<memN / 256, 256, 0, stream>>>(memory, hC, memN);
  cvt_f32_to_f16<<<xN / 256, 256, 0, stream>>>(x, hC + memN, xN);
  cvt_f32_to_f16<<<posN / 256, 256, 0, stream>>>(pos_emb, hP, posN);

  // 2) transpose+convert weights (B operands stored [N,K] so GEMMs are NT)
  wtrans_f16<<<(DM * HD) / 256, 256, 0, stream>>>(W_kv, hWkT, DM, HD, 2 * HD, 0);
  wtrans_f16<<<(DM * HD) / 256, 256, 0, stream>>>(W_kv, hWvT, DM, HD, 2 * HD, HD);
  wtrans_f16<<<(DM * HD) / 256, 256, 0, stream>>>(W_rel, hWrT, DM, HD, HD, 0);
  wtrans_f16<<<(DM * HD) / 256, 256, 0, stream>>>(W_q,  hWqT, DM, HD, HD, 0);
  wtrans_f16<<<(HD * DM) / 256, 256, 0, stream>>>(W_o,  hWoT, HD, DM, DM, 0);

  // 3) projection GEMMs (WMMA f16->f32, stored f16 where re-consumed by WMMA)
  // tiles = (M/32)*(N/64); 8 tiles per 256-thread block
  // K = c @ W_k : [8192,512]x[512,512] -> 2048 tiles
  gemm_nt_wmma<_Float16><<<dim3(256, 1), 256, 0, stream>>>(
      hC, hWkT, hK, TOT * BZ, HD, DM, DM, DM, HD, 0LL, 0LL, 0LL);
  // V = c @ W_v
  gemm_nt_wmma<_Float16><<<dim3(256, 1), 256, 0, stream>>>(
      hC, hWvT, hV, TOT * BZ, HD, DM, DM, DM, HD, 0LL, 0LL, 0LL);
  // rel = pos_emb @ W_rel
  gemm_nt_wmma<_Float16><<<dim3(256, 1), 256, 0, stream>>>(
      hP, hWrT, hRel, TOT * BZ, HD, DM, DM, DM, HD, 0LL, 0LL, 0LL);
  // q = x @ W_q (f32 out, bias added next); x f16 lives in tail of hC -> 512 tiles
  gemm_nt_wmma<float><<<dim3(64, 1), 256, 0, stream>>>(
      hC + memN, hWqT, fQ, LQ * BZ, HD, DM, DM, DM, HD, 0LL, 0LL, 0LL);

  // 4) qu = q + u, qv = q + v
  qbias_f16<<<xN / 256, 256, 0, stream>>>(fQ, pbu, pbv, hQu, hQv, xN);

  // 5) batched score GEMMs per (b,n): [512,64]x[64,2048] -> f32; 512 tiles/batch
  gemm_nt_wmma<float><<<dim3(64, BH), 256, 0, stream>>>(
      hQu, hK, fAC, LQ, TOT, DHD, BZ * HD, BZ * HD, TOT,
      (long long)DHD, (long long)DHD, (long long)LQ * TOT);
  gemm_nt_wmma<float><<<dim3(64, BH), 256, 0, stream>>>(
      hQv, hRel, fBD, LQ, TOT, DHD, BZ * HD, BZ * HD, TOT,
      (long long)DHD, (long long)DHD, (long long)LQ * TOT);

  // 6) fused rel-shift + gate + mask + softmax -> prob f16 [bn][i][c]
  softmax_kernel<<<dim3(LQ, BH), 256, 0, stream>>>(fAC, fBD, indice, hProb);

  // 7) attn_matrix = mean over (b,n)
  attn_mean_kernel<<<(LQ * TOT) / 256, 256, 0, stream>>>(hProb, out2);

  // 8) transpose V per (b,n) so PV GEMM is NT (contiguous along c)
  vtrans_f16<<<(BH * DHD * TOT) / 256, 256, 0, stream>>>(hV, hVT);

  // 9) vec = prob @ V : batched [512,2048]x[2048,64] -> f16 [i*B+b, HD]; 16 tiles/batch
  gemm_nt_wmma<_Float16><<<dim3(2, BH), 256, 0, stream>>>(
      hProb, hVT, hVec, LQ, DHD, TOT, TOT, TOT, BZ * HD,
      (long long)LQ * TOT, (long long)DHD * TOT, (long long)DHD);

  // 10) attn_out = vec @ W_o : [2048,512]x[512,512] -> f32; 512 tiles
  gemm_nt_wmma<float><<<dim3(64, 1), 256, 0, stream>>>(
      hVec, hWoT, fAO, LQ * BZ, DM, HD, HD, HD, DM, 0LL, 0LL, 0LL);

  // 11) residual + LayerNorm
  resid_ln_kernel<<<LQ * BZ, 256, 0, stream>>>(x, fAO, ln_g, ln_b, out1);
}
